// NN4_82910048682320
// MI455X (gfx1250) — compile-verified, matrix operand order checked by reading, not run on hardware
//
#include <hip/hip_runtime.h>
#include <cstdint>
#include <cstddef>

// ---------------------------------------------------------------------------
// MI455X / gfx1250 implementation.
// Convs -> explicit f16 im2col (memory-bound, HBM-rate) + pipelined implicit
// GEMM on v_wmma_f32_16x16x32_f16: double-buffered LDS tiles filled with
// gfx1250 async-to-LDS DMA (ASYNCcnt, in-order), fragments via ds_load_b128.
// BN / pool / small GEMVs / 3x3 complex MMSE solve are cheap VALU kernels.
// ---------------------------------------------------------------------------

typedef __attribute__((ext_vector_type(16))) _Float16 v16h;
typedef __attribute__((ext_vector_type(8)))  _Float16 v8h;
typedef __attribute__((ext_vector_type(8)))  float    v8f;

#define THREADS 256
#define CDIV(a,b) (((a)+(b)-1)/(b))

// --------------------------- f16 weight repack ------------------------------
// wgt[(CoutRows)][Kpad] f16, zero padded (extra rows let B-tile staging be
// uniform: every thread stages one 16B chunk with no guards).
__global__ void cvt_wgt_kernel(const float* __restrict__ w, _Float16* __restrict__ o,
                               int Cout, int K, int CoutRows, int Kpad) {
  int idx = blockIdx.x * blockDim.x + threadIdx.x;
  int total = CoutRows * Kpad;
  if (idx >= total) return;
  int n = idx / Kpad, k = idx - n * Kpad;
  float v = (n < Cout && k < K) ? w[(size_t)n * K + k] : 0.0f;
  o[idx] = (_Float16)v;
}

// --------------------------- f16 im2col -------------------------------------
// col[i][p][Kpad] f16, K = Cin*9 (3x3 SAME), zero halo + zero K-padding.
__global__ void im2col_f16_kernel(const float* __restrict__ in, _Float16* __restrict__ col,
                                  int Cin, int H, int W, int Kpad, int K, int total) {
  int idx = blockIdx.x * blockDim.x + threadIdx.x;
  if (idx >= total) return;
  int HW = H * W;
  int k = idx % Kpad;
  int p = (idx / Kpad) % HW;
  int i = idx / (Kpad * HW);
  float v = 0.0f;
  if (k < K) {
    int ci = k / 9;
    int rs = k - ci * 9;
    int r = rs / 3, s = rs - r * 3;
    int py = p / W, px = p - py * W;
    int iy = py + r - 1, ix = px + s - 1;
    if (iy >= 0 && iy < H && ix >= 0 && ix < W)
      v = in[((size_t)i * Cin + ci) * HW + iy * W + ix];
  }
  col[idx] = (_Float16)v;
}

// --------------------------- WMMA GEMM conv ---------------------------------
// Block: 256 threads = 8 waves; block tile = 128 pixels (M) x 16 couts (N).
// Per 32-K slice: A tile 128x32 f16 (8KB), B tile 64x32 f16 (4KB, 16 used),
// each wave issues exactly 3 async b128 DMAs -> s_wait_asynccnt 3 pipelines
// slice k+1 DMA under slice k WMMA (async loads complete in order).
__global__ __launch_bounds__(256)
void conv_gemm_wmma_kernel(const _Float16* __restrict__ col, const _Float16* __restrict__ wgt,
                           const float* __restrict__ bias, float* __restrict__ out,
                           int HW, int Cout, int Kpad) {
  __shared__ __align__(16) _Float16 sA[2][128 * 32];
  __shared__ __align__(16) _Float16 sB[2][64 * 32];

  const int tid  = threadIdx.x;
  const int lane = tid & 31;
  const int wave = tid >> 5;
  const int half = lane >> 4;
  const int ln   = lane & 15;
  const int img  = blockIdx.z;
  const int pixBase0 = blockIdx.x * 128;          // block M origin
  const int pixBase  = pixBase0 + wave * 16;      // wave M origin
  const int coutBase = blockIdx.y * 16;
  const size_t rowBase = (size_t)img * HW + pixBase0;  // first im2col row

  auto issue_slice = [&](int kc, int bufsel) {
    // A: 512 chunks of 16B, 2 per thread
    uint32_t abase = (uint32_t)(uintptr_t)(&sA[bufsel][0]);
#pragma unroll
    for (int c = 0; c < 2; ++c) {
      int chunk = tid + c * 256;
      int m = chunk >> 2, sub = chunk & 3;
      uint32_t goff = (uint32_t)(((rowBase + m) * Kpad + kc) * 2 + sub * 16);
      uint32_t ldst = abase + (uint32_t)(m * 64 + sub * 16);
      asm volatile("global_load_async_to_lds_b128 %0, %1, %2 offset:0"
                   :: "v"(ldst), "v"(goff), "s"(col) : "memory");
    }
    // B: 256 chunks of 16B, 1 per thread (rows 16..63 unused)
    uint32_t bbase = (uint32_t)(uintptr_t)(&sB[bufsel][0]);
    int n = tid >> 2, sub = tid & 3;
    uint32_t goff = (uint32_t)((((size_t)(coutBase + n)) * Kpad + kc) * 2 + sub * 16);
    uint32_t ldst = bbase + (uint32_t)(n * 64 + sub * 16);
    asm volatile("global_load_async_to_lds_b128 %0, %1, %2 offset:0"
                 :: "v"(ldst), "v"(goff), "s"(wgt) : "memory");
  };

  v8f acc = {};
  const int nslices = Kpad >> 5;
  issue_slice(0, 0);
  for (int s = 0; s < nslices; ++s) {
    int cur = s & 1;
    if (s + 1 < nslices) {
      issue_slice((s + 1) * 32, cur ^ 1);
      asm volatile("s_wait_asynccnt 3" ::: "memory");   // slice s landed (in-order)
    } else {
      asm volatile("s_wait_asynccnt 0" ::: "memory");
    }
    __syncthreads();

    // CDNA5 16-bit A/B layout: lane holds K = [8h..8h+7] and [16+8h..23+8h]
    // of its row -> two aligned b128 LDS loads per operand.
    const _Float16* aRow = &sA[cur][(wave * 16 + ln) * 32];
    const _Float16* bRow = &sB[cur][ln * 32];
    v8h a0 = *(const v8h*)(aRow + 8 * half);
    v8h a1 = *(const v8h*)(aRow + 16 + 8 * half);
    v8h b0 = *(const v8h*)(bRow + 8 * half);
    v8h b1 = *(const v8h*)(bRow + 16 + 8 * half);
    v16h a = __builtin_shufflevector(a0, a1, 0,1,2,3,4,5,6,7,8,9,10,11,12,13,14,15);
    v16h b = __builtin_shufflevector(b0, b1, 0,1,2,3,4,5,6,7,8,9,10,11,12,13,14,15);
    acc = __builtin_amdgcn_wmma_f32_16x16x32_f16(false, a, false, b,
                                                 (short)0, acc, false, false);
    __syncthreads();   // buffer cur is DMA target again next iteration
  }

  // C/D layout: N = lane%16 (cout), M = r + 8*half (pixel within wave tile)
  int co = coutBase + ln;
  if (co < Cout) {
    float bv = bias[co];
    float* outI = out + ((size_t)img * Cout + co) * HW;
#pragma unroll
    for (int r = 0; r < 8; ++r) {
      int pp = pixBase + r + 8 * half;
      if (pp < HW) outI[pp] = acc[r] + bv;
    }
  }
}

// --------------------------- BatchNorm2d (+ReLU) ----------------------------
__global__ void bn2d_stats_kernel(const float* __restrict__ x, float* __restrict__ mv,
                                  int I, int C, int HW) {
  __shared__ float ss[256], sq[256];
  int c = blockIdx.x;
  int n = I * HW;
  float s = 0.0f, s2 = 0.0f;
  for (int t = threadIdx.x; t < n; t += 256) {
    int i = t / HW, p = t - i * HW;
    float v = x[((size_t)i * C + c) * HW + p];
    s += v; s2 += v * v;
  }
  ss[threadIdx.x] = s; sq[threadIdx.x] = s2;
  __syncthreads();
  for (int o = 128; o > 0; o >>= 1) {
    if (threadIdx.x < o) { ss[threadIdx.x] += ss[threadIdx.x + o]; sq[threadIdx.x] += sq[threadIdx.x + o]; }
    __syncthreads();
  }
  if (threadIdx.x == 0) {
    float m = ss[0] / n;
    mv[2 * c] = m;
    mv[2 * c + 1] = sq[0] / n - m * m;
  }
}

__global__ void bn2d_apply_kernel(float* __restrict__ x, const float* __restrict__ mv,
                                  const float* __restrict__ g, const float* __restrict__ be,
                                  int I, int C, int HW) {
  int idx = blockIdx.x * blockDim.x + threadIdx.x;
  if (idx >= I * C * HW) return;
  int c = (idx / HW) % C;
  float y = g[c] * (x[idx] - mv[2 * c]) * rsqrtf(mv[2 * c + 1] + 1e-5f) + be[c];
  x[idx] = fmaxf(y, 0.0f);
}

// --------------------------- misc elementwise kernels -----------------------
__global__ void concat_ri_kernel(const float* __restrict__ xr, const float* __restrict__ xi,
                                 float* __restrict__ out, int I, int HW) {
  int idx = blockIdx.x * blockDim.x + threadIdx.x;
  if (idx >= I * 6 * HW) return;
  int p = idx % HW; int c = (idx / HW) % 6; int i = idx / (6 * HW);
  out[idx] = (c < 3) ? xr[((size_t)i * 3 + c) * HW + p]
                     : xi[((size_t)i * 3 + (c - 3)) * HW + p];
}

__global__ void maxpool2_kernel(const float* __restrict__ in, float* __restrict__ out,
                                int IC, int H, int W) {
  int Ho = H / 2, Wo = W / 2;
  int idx = blockIdx.x * blockDim.x + threadIdx.x;
  if (idx >= IC * Ho * Wo) return;
  int xo = idx % Wo; int yo = (idx / Wo) % Ho; int c = idx / (Wo * Ho);
  const float* b = in + (size_t)c * H * W + (2 * yo) * W + 2 * xo;
  out[idx] = fmaxf(fmaxf(b[0], b[1]), fmaxf(b[W], b[W + 1]));
}

__global__ void vout_kernel(const float* __restrict__ t4, const float* __restrict__ w,
                            const float* __restrict__ b, float* __restrict__ Vc,
                            int I, int HW) {
  int idx = blockIdx.x * blockDim.x + threadIdx.x;
  if (idx >= I * 3 * HW) return;
  int p = idx % HW; int k = (idx / HW) % 3; int i = idx / (3 * HW);
  float acc = b[k];
  for (int c = 0; c < 4; ++c) acc += w[k * 4 + c] * t4[((size_t)i * 4 + c) * HW + p];
  Vc[idx] = fminf(fmaxf(acc, 1e-4f), 1000.0f);
}

__global__ void linear_kernel(const float* __restrict__ in, const float* __restrict__ w,
                              const float* __restrict__ b, float* __restrict__ out,
                              int I, int K, int O) {
  int idx = blockIdx.x * blockDim.x + threadIdx.x;
  if (idx >= I * O) return;
  int o = idx % O, i = idx / O;
  float acc = b[o];
  for (int k = 0; k < K; ++k) acc += in[(size_t)i * K + k] * w[(size_t)k * O + o];
  out[idx] = acc;
}

__global__ void bn1d_relu_kernel(float* __restrict__ x, const float* __restrict__ g,
                                 const float* __restrict__ be, int I, int O) {
  int o = threadIdx.x;
  if (o >= O) return;
  float s = 0.0f, s2 = 0.0f;
  for (int i = 0; i < I; ++i) { float v = x[i * O + o]; s += v; s2 += v * v; }
  float m = s / I;
  float inv = rsqrtf(s2 / I - m * m + 1e-5f);
  for (int i = 0; i < I; ++i) {
    float v = g[o] * (x[i * O + o] - m) * inv + be[o];
    x[i * O + o] = fmaxf(v, 0.0f);
  }
}

__global__ void head_kernel(const float* __restrict__ a, const float* __restrict__ bb,
                            const float* __restrict__ h2w, const float* __restrict__ h2b,
                            const float* __restrict__ b2w, const float* __restrict__ b2b,
                            float* __restrict__ Hout, float* __restrict__ Rbout,
                            float* __restrict__ hws, float* __restrict__ s2ws, int I) {
  int i = blockIdx.x * blockDim.x + threadIdx.x;
  if (i >= I) return;
  float ang[3];
  for (int k = 0; k < 3; ++k) {
    float acc = h2b[k];
    for (int o = 0; o < 64; ++o) acc += a[i * 64 + o] * h2w[o * 3 + k];
    ang[k] = tanhf(acc);
  }
  float acc = b2b[0];
  for (int o = 0; o < 64; ++o) acc += bb[i * 64 + o] * b2w[o];
  float sb = expf(acc);
  float s2 = fminf(fmaxf(sb * sb, 1e-4f), 100.0f);
  s2ws[i] = s2;
  for (int m = 0; m < 3; ++m)
    for (int k = 0; k < 3; ++k) {
      float ph = 3.14159265358979323846f * (float)m * ang[k];
      float re = cosf(ph), im = sinf(ph);
      int o = ((i * 3 + m) * 3 + k) * 2;
      Hout[o] = re; Hout[o + 1] = im;
      hws[o] = re;  hws[o + 1] = im;
    }
  for (int m = 0; m < 3; ++m)
    for (int l = 0; l < 3; ++l) {
      int o = ((i * 3 + m) * 3 + l) * 2;
      Rbout[o] = (m == l) ? s2 : 0.0f;
      Rbout[o + 1] = 0.0f;
    }
}

// --------------------------- 3x3 complex MMSE solve -------------------------
struct cplx { float x, y; };
__device__ __forceinline__ cplx cmul(cplx a, cplx b) { return cplx{a.x*b.x - a.y*b.y, a.x*b.y + a.y*b.x}; }
__device__ __forceinline__ cplx cadd(cplx a, cplx b) { return cplx{a.x + b.x, a.y + b.y}; }
__device__ __forceinline__ cplx csub(cplx a, cplx b) { return cplx{a.x - b.x, a.y - b.y}; }
__device__ __forceinline__ cplx cconj(cplx a)        { return cplx{a.x, -a.y}; }
__device__ __forceinline__ cplx cdivc(cplx a, cplx b) {
  float d = b.x*b.x + b.y*b.y + 1e-30f;
  return cplx{(a.x*b.x + a.y*b.y) / d, (a.y*b.x - a.x*b.y) / d};
}

__global__ void solve_kernel(const float* __restrict__ xr, const float* __restrict__ xi,
                             const float* __restrict__ Vc, const float* __restrict__ hws,
                             const float* __restrict__ s2ws, float* __restrict__ sabs,
                             int I, int HW) {
  int idx = blockIdx.x * blockDim.x + threadIdx.x;
  if (idx >= I * HW) return;
  int p = idx % HW, i = idx / HW;
  cplx h[3][3];
  for (int m = 0; m < 3; ++m)
    for (int k = 0; k < 3; ++k) {
      int o = ((i * 3 + m) * 3 + k) * 2;
      h[m][k] = cplx{hws[o], hws[o + 1]};
    }
  float v[3];
  for (int k = 0; k < 3; ++k) v[k] = Vc[((size_t)i * 3 + k) * HW + p];
  float s2 = s2ws[i];
  cplx R[3][3];
  for (int m = 0; m < 3; ++m)
    for (int l = 0; l < 3; ++l) {
      cplx acc = cplx{(m == l) ? s2 : 0.0f, 0.0f};
      for (int k = 0; k < 3; ++k) {
        cplx t = cmul(h[m][k], cconj(h[l][k]));
        acc.x += v[k] * t.x; acc.y += v[k] * t.y;
      }
      R[m][l] = acc;
    }
  cplx adj[3][3];
  adj[0][0] = csub(cmul(R[1][1], R[2][2]), cmul(R[1][2], R[2][1]));
  adj[0][1] = csub(cmul(R[0][2], R[2][1]), cmul(R[0][1], R[2][2]));
  adj[0][2] = csub(cmul(R[0][1], R[1][2]), cmul(R[0][2], R[1][1]));
  adj[1][0] = csub(cmul(R[1][2], R[2][0]), cmul(R[1][0], R[2][2]));
  adj[1][1] = csub(cmul(R[0][0], R[2][2]), cmul(R[0][2], R[2][0]));
  adj[1][2] = csub(cmul(R[0][2], R[1][0]), cmul(R[0][0], R[1][2]));
  adj[2][0] = csub(cmul(R[1][0], R[2][1]), cmul(R[1][1], R[2][0]));
  adj[2][1] = csub(cmul(R[0][1], R[2][0]), cmul(R[0][0], R[2][1]));
  adj[2][2] = csub(cmul(R[0][0], R[1][1]), cmul(R[0][1], R[1][0]));
  cplx det = cadd(cadd(cmul(R[0][0], adj[0][0]), cmul(R[0][1], adj[1][0])),
                  cmul(R[0][2], adj[2][0]));
  cplx inv[3][3];
  for (int a2 = 0; a2 < 3; ++a2)
    for (int b2 = 0; b2 < 3; ++b2) inv[a2][b2] = cdivc(adj[a2][b2], det);
  cplx xv[3];
  for (int m = 0; m < 3; ++m)
    xv[m] = cplx{xr[((size_t)i * 3 + m) * HW + p], xi[((size_t)i * 3 + m) * HW + p]};
  for (int k = 0; k < 3; ++k) {
    cplx sh = cplx{0.0f, 0.0f};
    for (int m = 0; m < 3; ++m) {
      cplx Wk = cplx{0.0f, 0.0f};
      for (int l = 0; l < 3; ++l) {
        cplx t = cmul(cconj(h[l][k]), inv[l][m]);
        Wk.x += v[k] * t.x; Wk.y += v[k] * t.y;
      }
      cplx c = cmul(Wk, xv[m]);
      sh.x += c.x; sh.y += c.y;
    }
    sabs[(size_t)k * I * HW + idx] = sqrtf(sh.x * sh.x + sh.y * sh.y);
  }
}

// --------------------------- VAE reparam / decoder glue ---------------------
__global__ void reparam_kernel(const float* __restrict__ zz, const float* __restrict__ eps_k,
                               float* __restrict__ z, float* __restrict__ mu_out,
                               float* __restrict__ lv_out, int I, int dz) {
  int idx = blockIdx.x * blockDim.x + threadIdx.x;
  if (idx >= I * dz) return;
  int j = idx % dz, i = idx / dz;
  float mu = zz[i * 2 * dz + 2 * j];
  float lv = zz[i * 2 * dz + 2 * j + 1];
  z[idx] = mu + eps_k[idx] * expf(0.5f * lv);
  mu_out[idx] = mu;
  lv_out[idx] = lv;
}

__global__ void build_dec_in_kernel(float* __restrict__ out, const float* __restrict__ z,
                                    int I, int dz, int im) {
  int C = dz + 2, HW = im * im;
  int idx = blockIdx.x * blockDim.x + threadIdx.x;
  if (idx >= I * C * HW) return;
  int b = idx % im; int a = (idx / im) % im; int c = (idx / HW) % C; int i = idx / (HW * C);
  float v;
  if (c == 0)      v = -1.0f + 2.0f * (float)a / (float)(im - 1);
  else if (c == 1) v = -1.0f + 2.0f * (float)b / (float)(im - 1);
  else             v = z[i * dz + (c - 2)];
  out[idx] = v;
}

__global__ void decout_scatter_kernel(const float* __restrict__ d, const float* __restrict__ w,
                                      const float* __restrict__ b, float* __restrict__ vd,
                                      int I, int HW, int k) {
  int idx = blockIdx.x * blockDim.x + threadIdx.x;
  if (idx >= I * HW) return;
  int p = idx % HW, i = idx / HW;
  float acc = b[0];
  for (int c = 0; c < 4; ++c) acc += w[c] * d[((size_t)i * 4 + c) * HW + p];
  float vi = fminf(fmaxf(expf(acc), 1e-4f), 1000.0f);
  size_t base = (((size_t)idx * 3 + (size_t)k) * 3) * 2;   // vd[i, a, b, k, :] complex
  for (int k2 = 0; k2 < 3; ++k2) {
    vd[base + 2 * k2]     = (k2 == k) ? vi : 0.0f;
    vd[base + 2 * k2 + 1] = 0.0f;
  }
}

// --------------------------- host orchestration -----------------------------
static void launch_conv3x3(const float* in, const float* w, const float* bias, float* out,
                           int I, int Cin, int H, int W, int Cout,
                           _Float16* wgt, _Float16* col, hipStream_t stream) {
  int K = Cin * 9;
  int Kpad = (K + 31) & ~31;
  int CoutPad = (Cout + 15) & ~15;
  int HW = H * W;
  int CoutRows = CoutPad + 48;   // zero slack so B staging needs no guards
  cvt_wgt_kernel<<<CDIV(CoutRows * Kpad, THREADS), THREADS, 0, stream>>>(w, wgt, Cout, K, CoutRows, Kpad);
  int tot = I * HW * Kpad;       // max 32*10000*576 = 184.3M, fits int32
  im2col_f16_kernel<<<CDIV(tot, THREADS), THREADS, 0, stream>>>(in, col, Cin, H, W, Kpad, K, tot);
  dim3 g(CDIV(HW, 128), CoutPad / 16, I);
  conv_gemm_wmma_kernel<<<g, 256, 0, stream>>>(col, wgt, bias, out, HW, Cout, Kpad);
}

static void launch_bn_relu(float* x, const float* g, const float* be,
                           int I, int C, int HW, float* mv, hipStream_t stream) {
  bn2d_stats_kernel<<<C, 256, 0, stream>>>(x, mv, I, C, HW);
  bn2d_apply_kernel<<<CDIV(I * C * HW, THREADS), THREADS, 0, stream>>>(x, mv, g, be, I, C, HW);
}

struct DC { const float *b1, *b2, *be1, *be2, *g1, *g2, *w1, *w2; };
static DC dc_at(void* const* d, int base) {
  DC p;
  p.b1  = (const float*)d[base + 0];  p.b2  = (const float*)d[base + 1];
  p.be1 = (const float*)d[base + 2];  p.be2 = (const float*)d[base + 3];
  p.g1  = (const float*)d[base + 4];  p.g2  = (const float*)d[base + 5];
  p.w1  = (const float*)d[base + 6];  p.w2  = (const float*)d[base + 7];
  return p;
}

static void run_dc(const float* in, float* out, float* tmp, const DC& p,
                   int I, int Cin, int Cout, int H, int W,
                   _Float16* wgt, _Float16* col, float* mv, hipStream_t stream) {
  launch_conv3x3(in,  p.w1, p.b1, tmp, I, Cin,  H, W, Cout, wgt, col, stream);
  launch_bn_relu(tmp, p.g1, p.be1, I, Cout, H * W, mv, stream);
  launch_conv3x3(tmp, p.w2, p.b2, out, I, Cout, H, W, Cout, wgt, col, stream);
  launch_bn_relu(out, p.g2, p.be2, I, Cout, H * W, mv, stream);
}

extern "C" void kernel_launch(void* const* d_in, const int* in_sizes, int n_in,
                              void* d_out, int out_size, void* d_ws, size_t ws_size,
                              hipStream_t stream) {
  (void)in_sizes; (void)n_in; (void)out_size; (void)ws_size;
  const int I = 32, HW = 10000, dz = 32, im = 100;

  // ---- inputs (jax pytree flatten: dict keys sorted alphabetically) --------
  const float* xr  = (const float*)d_in[0];
  const float* xi  = (const float*)d_in[1];
  const float* eps = (const float*)d_in[2];
  const float* b1_b = (const float*)d_in[3];
  const float* b1_w = (const float*)d_in[4];
  const float* b2_b = (const float*)d_in[5];
  const float* b2_w = (const float*)d_in[6];
  const float* bbn_be = (const float*)d_in[7];
  const float* bbn_g  = (const float*)d_in[8];
  DC dec0 = dc_at(d_in, 9),  dec1 = dc_at(d_in, 17);
  DC dec2 = dc_at(d_in, 25), dec3 = dc_at(d_in, 33);
  const float* dout_b = (const float*)d_in[41];
  const float* dout_w = (const float*)d_in[42];
  DC enc0 = dc_at(d_in, 43), enc1 = dc_at(d_in, 51);
  DC enc2 = dc_at(d_in, 59), enc3 = dc_at(d_in, 67);
  const float* fc1_b = (const float*)d_in[75];
  const float* fc1_w = (const float*)d_in[76];
  const float* h1_b = (const float*)d_in[77];
  const float* h1_w = (const float*)d_in[78];
  const float* h2_b = (const float*)d_in[79];
  const float* h2_w = (const float*)d_in[80];
  const float* hbn_be = (const float*)d_in[81];
  const float* hbn_g  = (const float*)d_in[82];
  DC hb0 = dc_at(d_in, 83), hb1 = dc_at(d_in, 91), hb2 = dc_at(d_in, 99);
  DC vn0 = dc_at(d_in, 107), vn1 = dc_at(d_in, 115), vn2 = dc_at(d_in, 123);
  const float* vo_b = (const float*)d_in[131];
  const float* vo_w = (const float*)d_in[132];

  // ---- outputs: (vd, Hhat, Rb, mu, logvar) flat, complex64 as float pairs --
  float* vd   = (float*)d_out;          // 32*100*100*3*3 complex = 5,760,000 f
  float* Hout = vd + 5760000;           // 576 f
  float* Rbo  = Hout + 576;             // 576 f
  float* muo  = Rbo + 576;              // 1024 f
  float* lvo  = muo + 1024;             // 1024 f

  // ---- workspace carve -----------------------------------------------------
  char* wsb = (char*)d_ws;
  size_t off = 0;
  auto carve = [&](size_t bytes) -> void* {
    void* p = wsb + off;
    off = (off + bytes + 255) & ~(size_t)255;
    return p;
  };
  _Float16* wgt = (_Float16*)carve((size_t)(64 + 48) * 576 * sizeof(_Float16));
  // im2col: max I*HW*Kpad = 32*10000*576 f16, + 128-row slack for last M-tile
  _Float16* col = (_Float16*)carve(((size_t)I * HW + 128) * 576 * sizeof(_Float16));
  float* bufA = (float*)carve((size_t)I * 64 * HW * sizeof(float));
  float* bufB = (float*)carve((size_t)I * 64 * HW * sizeof(float));
  float* t4   = (float*)carve((size_t)I * 4 * HW * sizeof(float));
  float* Vc   = (float*)carve((size_t)I * 3 * HW * sizeof(float));
  float* sabs = (float*)carve((size_t)3 * I * HW * sizeof(float));
  float* mv   = (float*)carve(2 * 64 * sizeof(float));
  float* hws  = (float*)carve(576 * sizeof(float));
  float* s2ws = (float*)carve(I * sizeof(float));
  float* a64  = (float*)carve(I * 64 * sizeof(float));
  float* bb64 = (float*)carve(I * 64 * sizeof(float));
  float* zz   = (float*)carve(I * 64 * sizeof(float));
  float* zbuf = (float*)carve(I * dz * sizeof(float));

  // ---- v_net: t = concat(xr,xi) -> 3x dc ----------------------------------
  concat_ri_kernel<<<CDIV(I * 6 * HW, THREADS), THREADS, 0, stream>>>(xr, xi, bufA, I, HW);
  run_dc(bufA, bufA, bufB, vn0, I, 6, 32, 100, 100, wgt, col, mv, stream);
  run_dc(bufA, bufA, bufB, vn1, I, 32, 16, 100, 100, wgt, col, mv, stream);
  run_dc(bufA, t4,   bufB, vn2, I, 16, 4, 100, 100, wgt, col, mv, stream);
  vout_kernel<<<CDIV(I * 3 * HW, THREADS), THREADS, 0, stream>>>(t4, vo_w, vo_b, Vc, I, HW);

  // ---- hb_net: 3x (pool -> dc) --------------------------------------------
  maxpool2_kernel<<<CDIV(I * 4 * 2500, THREADS), THREADS, 0, stream>>>(t4, bufA, I * 4, 100, 100);
  run_dc(bufA, bufA, bufB, hb0, I, 4, 32, 50, 50, wgt, col, mv, stream);
  maxpool2_kernel<<<CDIV(I * 32 * 625, THREADS), THREADS, 0, stream>>>(bufA, bufB, I * 32, 50, 50);
  run_dc(bufB, bufB, bufA, hb1, I, 32, 16, 25, 25, wgt, col, mv, stream);
  maxpool2_kernel<<<CDIV(I * 16 * 144, THREADS), THREADS, 0, stream>>>(bufB, bufA, I * 16, 25, 25);
  run_dc(bufA, bufA, bufB, hb2, I, 16, 8, 12, 12, wgt, col, mv, stream);  // hb = bufA (I x 1152)

  // ---- heads: ang / sig_b -> Hhat, Rb -------------------------------------
  linear_kernel<<<CDIV(I * 64, THREADS), THREADS, 0, stream>>>(bufA, h1_w, h1_b, a64, I, 1152, 64);
  bn1d_relu_kernel<<<1, 64, 0, stream>>>(a64, hbn_g, hbn_be, I, 64);
  linear_kernel<<<CDIV(I * 64, THREADS), THREADS, 0, stream>>>(bufA, b1_w, b1_b, bb64, I, 1152, 64);
  bn1d_relu_kernel<<<1, 64, 0, stream>>>(bb64, bbn_g, bbn_be, I, 64);
  head_kernel<<<1, 32, 0, stream>>>(a64, bb64, h2_w, h2_b, b2_w, b2_b, Hout, Rbo, hws, s2ws, I);

  // ---- per-pixel 3x3 complex MMSE: shat magnitudes ------------------------
  solve_kernel<<<CDIV(I * HW, THREADS), THREADS, 0, stream>>>(xr, xi, Vc, hws, s2ws, sabs, I, HW);

  // ---- per-source encoder/decoder VAE passes ------------------------------
  for (int k = 0; k < 3; ++k) {
    const float* s = sabs + (size_t)k * I * HW;
    maxpool2_kernel<<<CDIV(I * 2500, THREADS), THREADS, 0, stream>>>(s, bufA, I, 100, 100);
    run_dc(bufA, bufA, bufB, enc0, I, 1, 64, 50, 50, wgt, col, mv, stream);
    run_dc(bufA, bufA, bufB, enc1, I, 64, 32, 50, 50, wgt, col, mv, stream);
    maxpool2_kernel<<<CDIV(I * 32 * 625, THREADS), THREADS, 0, stream>>>(bufA, bufB, I * 32, 50, 50);
    run_dc(bufB, bufB, bufA, enc2, I, 32, 16, 25, 25, wgt, col, mv, stream);
    run_dc(bufB, bufB, bufA, enc3, I, 16, 1, 25, 25, wgt, col, mv, stream);  // (I, 625)
    linear_kernel<<<CDIV(I * 64, THREADS), THREADS, 0, stream>>>(bufB, fc1_w, fc1_b, zz, I, 625, 64);
    reparam_kernel<<<CDIV(I * dz, THREADS), THREADS, 0, stream>>>(zz, eps + (size_t)k * I * dz,
                                                                  zbuf, muo, lvo, I, dz);
    build_dec_in_kernel<<<CDIV(I * (dz + 2) * HW, THREADS), THREADS, 0, stream>>>(bufA, zbuf, I, dz, im);
    run_dc(bufA, bufA, bufB, dec0, I, 34, 64, 100, 100, wgt, col, mv, stream);
    run_dc(bufA, bufA, bufB, dec1, I, 64, 32, 100, 100, wgt, col, mv, stream);
    run_dc(bufA, bufA, bufB, dec2, I, 32, 16, 100, 100, wgt, col, mv, stream);
    run_dc(bufA, bufA, bufB, dec3, I, 16, 4, 100, 100, wgt, col, mv, stream);
    decout_scatter_kernel<<<CDIV(I * HW, THREADS), THREADS, 0, stream>>>(bufA, dout_w, dout_b, vd, I, HW, k);
  }
}
